// AttentionHeadless_58196806861322
// MI455X (gfx1250) — compile-verified, matrix operand order checked by reading, not run on hardware
//
#include <hip/hip_runtime.h>
#include <hip/hip_bf16.h>

// B=2, Q=512, K=512, T=256
// out[b,q,u] = sum_t value_sum[b,q,t] * Wvo[u,t]
// value_sum[b,q,u] = sum_k vproj[b,k,u] * scale[b,k,q,u]
// scale = swishmax over Q of logits[b,k,q,u] = sum_t qproj[b,q,t]*kproj[b,k,t]*Wal[u,t]

typedef __bf16 bf16_t;
typedef __attribute__((ext_vector_type(16))) __bf16 v16bf;
typedef __attribute__((ext_vector_type(8)))  __bf16 v8bf;
typedef __attribute__((ext_vector_type(8)))  float  v8f;
typedef unsigned int u32x4 __attribute__((ext_vector_type(4)));
typedef int          i32x4 __attribute__((ext_vector_type(4)));
typedef int          i32x8 __attribute__((ext_vector_type(8)));

constexpr int Bdim = 2;
constexpr int Qdim = 512;
constexpr int Kdim = 512;
constexpr int Tdim = 256;

#ifndef __has_builtin
#define __has_builtin(x) 0
#endif
#if __has_builtin(__builtin_amdgcn_tensor_load_to_lds) && \
    __has_builtin(__builtin_amdgcn_s_wait_tensorcnt)
#define USE_TDM 1
#else
#define USE_TDM 0
#endif

// ---------------- Tensor Data Mover: 1-D contiguous copy global -> LDS ----------------
// D# per CDNA5 ISA: group0 {count=1, lds_addr, global_addr, type=2},
// group1 {data_size=8B, tensor_dim0=tile_dim0=nelem8, stride0=nelem8}. Issued by one wave;
// completion via TENSORcnt.
__device__ __forceinline__ void tdm_load_1d(unsigned lds_byte_off, const void* src,
                                            unsigned nelem8) {
#if USE_TDM
    unsigned long long ga = (unsigned long long)src;
    u32x4 g0 = {1u,                                   // count=1 (valid), user mode
                lds_byte_off,                         // lds_addr (bytes)
                (unsigned)ga,                         // global_addr[31:0]
                (unsigned)((ga >> 32) & 0x01FFFFFFu) | 0x80000000u}; // [56:32] | type=2
    i32x8 g1 = {(int)(3u << 16),                      // data_size = 8 bytes, no flags
                (int)((nelem8 & 0xFFFFu) << 16),      // tensor_dim0[15:0] @ bits63:48
                (int)((nelem8 >> 16) & 0xFFFFu),      // tensor_dim0[31:16]; tensor_dim1=0
                (int)((nelem8 & 0xFFFFu) << 16),      // tile_dim0 @ bits127:112
                0,                                    // tile_dim1=0, tile_dim2=0 (unused)
                (int)nelem8,                          // tensor_dim0_stride[31:0]
                0, 0};
    i32x4 z4 = {0, 0, 0, 0};
#if __clang_major__ >= 23
    i32x8 z8 = {0, 0, 0, 0, 0, 0, 0, 0};
    __builtin_amdgcn_tensor_load_to_lds(g0, g1, z4, z4, z8, 0);
#else
    __builtin_amdgcn_tensor_load_to_lds(g0, g1, z4, z4, 0);
#endif
#endif
}

__device__ __forceinline__ void tdm_wait() {
#if USE_TDM
    __builtin_amdgcn_s_wait_tensorcnt(0);
#endif
}

// ---------------- fragment loaders (16-bit A layout; off0 includes half-lane shift) -----
__device__ __forceinline__ v16bf load_frag_bf(const bf16_t* __restrict__ row, int off0) {
    const v8bf lo = *(const v8bf*)(row + off0);
    const v8bf hi = *(const v8bf*)(row + off0 + 16);
    v16bf r;
#pragma unroll
    for (int i = 0; i < 8; ++i) { r[i] = lo[i]; r[i + 8] = hi[i]; }
    return r;
}

// elementwise bf16 product of two rows (folds kproj row into Wal columns) -> v_pk_mul_bf16
__device__ __forceinline__ v16bf load_frag_qk(const bf16_t* __restrict__ arow,
                                              const bf16_t* __restrict__ brow, int off0) {
    const v8bf p0 = *(const v8bf*)(arow + off0)      * *(const v8bf*)(brow + off0);
    const v8bf p1 = *(const v8bf*)(arow + off0 + 16) * *(const v8bf*)(brow + off0 + 16);
    v16bf r;
#pragma unroll
    for (int i = 0; i < 8; ++i) { r[i] = p0[i]; r[i + 8] = p1[i]; }
    return r;
}

__device__ __forceinline__ v16bf load_frag_f32(const float* __restrict__ row, int off0) {
    const float4 a0 = *(const float4*)(row + off0);
    const float4 a1 = *(const float4*)(row + off0 + 4);
    const float4 b0 = *(const float4*)(row + off0 + 16);
    const float4 b1 = *(const float4*)(row + off0 + 20);
    v16bf r;
    r[0]  = (bf16_t)a0.x; r[1]  = (bf16_t)a0.y; r[2]  = (bf16_t)a0.z; r[3]  = (bf16_t)a0.w;
    r[4]  = (bf16_t)a1.x; r[5]  = (bf16_t)a1.y; r[6]  = (bf16_t)a1.z; r[7]  = (bf16_t)a1.w;
    r[8]  = (bf16_t)b0.x; r[9]  = (bf16_t)b0.y; r[10] = (bf16_t)b0.z; r[11] = (bf16_t)b0.w;
    r[12] = (bf16_t)b1.x; r[13] = (bf16_t)b1.y; r[14] = (bf16_t)b1.z; r[15] = (bf16_t)b1.w;
    return r;
}

// ---------------- kernel 0: weights -> bf16, zero the k-reduction buffer ----------------
__global__ __launch_bounds__(256) void prep_kernel(
    const float* __restrict__ Wk, const float* __restrict__ Wq, const float* __restrict__ Wva,
    const float* __restrict__ Wal, const float* __restrict__ Wvo,
    bf16_t* __restrict__ wkb, bf16_t* __restrict__ wqb, bf16_t* __restrict__ wvab,
    bf16_t* __restrict__ walb, bf16_t* __restrict__ wvob, float* __restrict__ vsum) {
    const int stride = gridDim.x * blockDim.x;
    const int t0 = blockIdx.x * blockDim.x + threadIdx.x;
    for (int i = t0; i < Tdim * Tdim; i += stride) {
        wkb[i]  = (bf16_t)Wk[i];
        wqb[i]  = (bf16_t)Wq[i];
        wvab[i] = (bf16_t)Wva[i];
        walb[i] = (bf16_t)Wal[i];
        wvob[i] = (bf16_t)Wvo[i];
    }
    for (int i = t0; i < Bdim * Qdim * Tdim; i += stride) vsum[i] = 0.0f;
}

// ---------------- generic GEMM: out[m,u] = sum_t X[m,t] * W[u,t]; M = 1024 ----------------
__global__ __launch_bounds__(256) void gemm_xwT(const float* __restrict__ X,
                                                const bf16_t* __restrict__ Wb,
                                                void* __restrict__ outv, int out_bf16) {
    extern __shared__ char smem[];
    bf16_t* ldsw = (bf16_t*)smem; // 256x256 bf16 = 128KB @ LDS offset 0
#if USE_TDM
    if (threadIdx.x == 0) tdm_load_1d(0u, Wb, Tdim * Tdim / 4);
    tdm_wait();
#else
    {
        const float4* s = (const float4*)Wb;
        float4* d = (float4*)ldsw;
        for (int i = threadIdx.x; i < (Tdim * Tdim * 2) / 16; i += 256) d[i] = s[i];
    }
#endif
    __syncthreads();

    const int lane  = threadIdx.x & 31;
    const int wid   = threadIdx.x >> 5;
    const int lrow  = lane & 15;
    const int coff  = (lane & 16) ? 8 : 0;
    const int rhalf = (lane & 16) ? 8 : 0;
    const int mBase = blockIdx.x * 64;

    for (int j = 0; j < 2; ++j) {
        const int ub = wid * 32 + j * 16;
        const bf16_t* wrow = ldsw + (size_t)(ub + lrow) * Tdim;
        v16bf Bf[8];
#pragma unroll
        for (int kk = 0; kk < 8; ++kk) Bf[kk] = load_frag_bf(wrow, kk * 32 + coff);

        for (int ms = 0; ms < 4; ms += 2) {  // two independent accumulator chains
            const float* xrow0 = X + (size_t)(mBase + ms * 16 + lrow) * Tdim;
            const float* xrow1 = xrow0 + (size_t)16 * Tdim;
            v8f c0 = {}, c1 = {};
#pragma unroll
            for (int kk = 0; kk < 8; ++kk) {
                c0 = __builtin_amdgcn_wmma_f32_16x16x32_bf16(
                    false, load_frag_f32(xrow0, kk * 32 + coff), false, Bf[kk],
                    (short)0, c0, false, false);
                c1 = __builtin_amdgcn_wmma_f32_16x16x32_bf16(
                    false, load_frag_f32(xrow1, kk * 32 + coff), false, Bf[kk],
                    (short)0, c1, false, false);
            }
            const int col = ub + lrow;
            const int row0 = mBase + ms * 16 + rhalf;
            if (out_bf16) {
                bf16_t* o = (bf16_t*)outv;
#pragma unroll
                for (int r = 0; r < 8; ++r) {
                    o[(size_t)(row0 + r) * Tdim + col]      = (bf16_t)c0[r];
                    o[(size_t)(row0 + 16 + r) * Tdim + col] = (bf16_t)c1[r];
                }
            } else {
                float* o = (float*)outv;
#pragma unroll
                for (int r = 0; r < 8; ++r) {
                    o[(size_t)(row0 + r) * Tdim + col]      = c0[r];
                    o[(size_t)(row0 + 16 + r) * Tdim + col] = c1[r];
                }
            }
        }
    }
}

// ---------------- pass 1: per (b,k) online swishmax stats over Q ----------------
__global__ __launch_bounds__(256) void attn_stats(const bf16_t* __restrict__ qproj,
                                                  const bf16_t* __restrict__ kproj,
                                                  const bf16_t* __restrict__ walb,
                                                  float* __restrict__ stats_m,
                                                  float* __restrict__ stats_s) {
    extern __shared__ char smem[];
    bf16_t* ldsw = (bf16_t*)smem;             // Wal bf16, 128KB @ 0
    bf16_t* ldsk = (bf16_t*)(smem + 131072);  // kproj row, 512B @ 131072
    const int bk = blockIdx.x;                // b*K + k
    const int b  = bk >> 9;
#if USE_TDM
    if (threadIdx.x == 0) {
        tdm_load_1d(0u, walb, Tdim * Tdim / 4);
        tdm_load_1d(131072u, kproj + (size_t)bk * Tdim, Tdim / 4);
    }
    tdm_wait();
#else
    {
        const float4* s = (const float4*)walb;
        float4* d = (float4*)ldsw;
        for (int i = threadIdx.x; i < 8192; i += 256) d[i] = s[i];
        if (threadIdx.x < 32)
            ((float4*)ldsk)[threadIdx.x] =
                ((const float4*)(kproj + (size_t)bk * Tdim))[threadIdx.x];
    }
#endif
    __syncthreads();

    const int lane = threadIdx.x & 31;
    const int wid  = threadIdx.x >> 5;
    const int lrow = lane & 15;
    const int coff = (lane & 16) ? 8 : 0;
    const bf16_t* qb = qproj + (size_t)b * Qdim * Tdim;

    for (int j = 0; j < 2; ++j) {
        const int ub = wid * 32 + j * 16;
        const bf16_t* wrow = ldsw + (size_t)(ub + lrow) * Tdim;
        v16bf Bs[8]; // Wal columns pre-scaled by kproj row (v_pk_mul_bf16)
#pragma unroll
        for (int kk = 0; kk < 8; ++kk) Bs[kk] = load_frag_qk(wrow, ldsk, kk * 32 + coff);

        float m_run = -1e30f, s_run = 0.0f;
        for (int q0 = 0; q0 < Qdim; q0 += 32) {  // two 16-row tiles per iteration
            const bf16_t* qrow0 = qb + (size_t)(q0 + lrow) * Tdim;
            const bf16_t* qrow1 = qrow0 + (size_t)16 * Tdim;
            v8f c0 = {}, c1 = {};
#pragma unroll
            for (int kk = 0; kk < 8; ++kk) {
                c0 = __builtin_amdgcn_wmma_f32_16x16x32_bf16(
                    false, load_frag_bf(qrow0, kk * 32 + coff), false, Bs[kk],
                    (short)0, c0, false, false);
                c1 = __builtin_amdgcn_wmma_f32_16x16x32_bf16(
                    false, load_frag_bf(qrow1, kk * 32 + coff), false, Bs[kk],
                    (short)0, c1, false, false);
            }
            float tmax = c0[0];
#pragma unroll
            for (int r = 1; r < 8; ++r) tmax = fmaxf(tmax, c0[r]);
#pragma unroll
            for (int r = 0; r < 8; ++r) tmax = fmaxf(tmax, c1[r]);
            const float newm = fmaxf(m_run, tmax);
            s_run *= __expf(m_run - newm);
#pragma unroll
            for (int r = 0; r < 8; ++r) s_run += fabsf(c0[r] * __expf(c0[r] - newm));
#pragma unroll
            for (int r = 0; r < 8; ++r) s_run += fabsf(c1[r] * __expf(c1[r] - newm));
            m_run = newm;
        }
        // merge the two half-lane partials (same u, other 8-row half)
        const float mo = __shfl_xor(m_run, 16, 32);
        const float so = __shfl_xor(s_run, 16, 32);
        const float mf = fmaxf(m_run, mo);
        const float sf = s_run * __expf(m_run - mf) + so * __expf(mo - mf);
        if (lane < 16) {
            const size_t base = (size_t)bk * Tdim + ub + lane;
            stats_m[base] = mf;
            stats_s[base] = sf;
        }
    }
}

// ---------------- pass 2: recompute logits, apply scale*v, reduce over k ----------------
__global__ __launch_bounds__(256) void attn_accum(const bf16_t* __restrict__ qproj,
                                                  const bf16_t* __restrict__ kproj,
                                                  const bf16_t* __restrict__ walb,
                                                  const float* __restrict__ vproj,
                                                  const float* __restrict__ stats_m,
                                                  const float* __restrict__ stats_s,
                                                  float* __restrict__ vsum) {
    extern __shared__ char smem[];
    bf16_t* ldsw = (bf16_t*)smem;                      // Wal: 128KB @ 0
    bf16_t* ldsk = (bf16_t*)(smem + 131072);           // 32 kproj rows: 16KB @ 131072
    bf16_t* ldsq = (bf16_t*)(smem + 131072 + 16384);   // 64 qproj rows: 32KB @ 147456
    const int gid = blockIdx.x;
    const int b   = gid >> 7;
    const int rem = gid & 127;
    const int qt  = rem >> 4;        // q-tile of 64 rows
    const int k0  = (rem & 15) * 32; // k-chunk of 32
#if USE_TDM
    if (threadIdx.x == 0) {
        tdm_load_1d(0u, walb, Tdim * Tdim / 4);
        tdm_load_1d(131072u, kproj + ((size_t)b * Kdim + k0) * Tdim, 32 * Tdim / 4);
        tdm_load_1d(147456u, qproj + ((size_t)b * Qdim + qt * 64) * Tdim, 64 * Tdim / 4);
    }
    tdm_wait();
#else
    {
        const float4* s = (const float4*)walb;
        float4* d = (float4*)ldsw;
        for (int i = threadIdx.x; i < 8192; i += 256) d[i] = s[i];
        const float4* ks = (const float4*)(kproj + ((size_t)b * Kdim + k0) * Tdim);
        float4* kd = (float4*)ldsk;
        for (int i = threadIdx.x; i < 1024; i += 256) kd[i] = ks[i];
        const float4* qs = (const float4*)(qproj + ((size_t)b * Qdim + qt * 64) * Tdim);
        float4* qd = (float4*)ldsq;
        for (int i = threadIdx.x; i < 2048; i += 256) qd[i] = qs[i];
    }
#endif
    __syncthreads();

    const int lane  = threadIdx.x & 31;
    const int wid   = threadIdx.x >> 5;
    const int lrow  = lane & 15;
    const int coff  = (lane & 16) ? 8 : 0;
    const int rhalf = (lane & 16) ? 8 : 0;

    for (int j = 0; j < 2; ++j) {
        const int ub = wid * 32 + j * 16;
        const bf16_t* wrow = ldsw + (size_t)(ub + lrow) * Tdim;
        v8f acc[4] = {};
        for (int kk2 = 0; kk2 < 32; ++kk2) {
            const int k = k0 + kk2;
            const size_t sbase = ((size_t)b * Kdim + k) * Tdim + ub + lrow;
            const float mv  = stats_m[sbase];
            const float sv  = stats_s[sbase];
            const float vv  = vproj[sbase];
            const float inv = vv / (sv + 1.0f); // NOT_EPSILON = 1.0
            const bf16_t* krow = ldsk + (size_t)kk2 * Tdim;
            v16bf Bs[8];
#pragma unroll
            for (int kk = 0; kk < 8; ++kk) Bs[kk] = load_frag_qk(wrow, krow, kk * 32 + coff);
#pragma unroll
            for (int msp = 0; msp < 4; msp += 2) {  // two independent chains
                const bf16_t* qrow0 = ldsq + (size_t)(msp * 16 + lrow) * Tdim;
                const bf16_t* qrow1 = qrow0 + (size_t)16 * Tdim;
                v8f c0 = {}, c1 = {};
#pragma unroll
                for (int kk = 0; kk < 8; ++kk) {
                    c0 = __builtin_amdgcn_wmma_f32_16x16x32_bf16(
                        false, load_frag_bf(qrow0, kk * 32 + coff), false, Bs[kk],
                        (short)0, c0, false, false);
                    c1 = __builtin_amdgcn_wmma_f32_16x16x32_bf16(
                        false, load_frag_bf(qrow1, kk * 32 + coff), false, Bs[kk],
                        (short)0, c1, false, false);
                }
#pragma unroll
                for (int r = 0; r < 8; ++r) {
                    acc[msp][r]     += c0[r] * __expf(c0[r] - mv) * inv;
                    acc[msp + 1][r] += c1[r] * __expf(c1[r] - mv) * inv;
                }
            }
            if (kk2 + 1 < 32) __builtin_prefetch(stats_m + sbase + Tdim, 0, 0);
        }
        const int col = ub + lrow;
#pragma unroll
        for (int ms = 0; ms < 4; ++ms)
#pragma unroll
            for (int r = 0; r < 8; ++r) {
                const int row = qt * 64 + ms * 16 + rhalf + r;
                atomicAdd(&vsum[((size_t)b * Qdim + row) * Tdim + col], acc[ms][r]);
            }
    }
}

// ---------------- launch ----------------
extern "C" void kernel_launch(void* const* d_in, const int* in_sizes, int n_in,
                              void* d_out, int out_size, void* d_ws, size_t ws_size,
                              hipStream_t stream) {
    const float* query = (const float*)d_in[0]; // [B,Q,T]
    const float* keyt  = (const float*)d_in[1]; // [B,K,T]
    const float* valt  = (const float*)d_in[2]; // [B,K,T]
    const float* Wk    = (const float*)d_in[3];
    const float* Wq    = (const float*)d_in[4];
    const float* Wva   = (const float*)d_in[5];
    const float* Wal   = (const float*)d_in[6];
    const float* Wvo   = (const float*)d_in[7];

    char* p = (char*)d_ws;
    bf16_t* wkb     = (bf16_t*)(p + 0);        // 128KB each
    bf16_t* wqb     = (bf16_t*)(p + 131072);
    bf16_t* wvab    = (bf16_t*)(p + 262144);
    bf16_t* walb    = (bf16_t*)(p + 393216);
    bf16_t* wvob    = (bf16_t*)(p + 524288);
    bf16_t* kprojb  = (bf16_t*)(p + 655360);   // [B,K,T] bf16: 512KB
    bf16_t* qprojb  = (bf16_t*)(p + 1179648);  // [B,Q,T] bf16: 512KB
    float*  vproj   = (float*)(p + 1703936);   // [B,K,T] f32: 1MB
    float*  stats_m = (float*)(p + 2752512);   // [B,K,T] f32: 1MB
    float*  stats_s = (float*)(p + 3801088);   // [B,K,T] f32: 1MB
    float*  vsum    = (float*)(p + 4849664);   // [B,Q,T] f32: 1MB  (total ~5.6MB)

    prep_kernel<<<256, 256, 0, stream>>>(Wk, Wq, Wva, Wal, Wvo,
                                         wkb, wqb, wvab, walb, wvob, vsum);

    // projections: y = x @ W^T  (M = B*512 = 1024 rows each)
    gemm_xwT<<<16, 256, 131072, stream>>>(keyt,  wkb,  (void*)kprojb, 1);
    gemm_xwT<<<16, 256, 131072, stream>>>(query, wqb,  (void*)qprojb, 1);
    gemm_xwT<<<16, 256, 131072, stream>>>(valt,  wvab, (void*)vproj,  0);

    attn_stats<<<Bdim * Kdim, 256, 131584, stream>>>(qprojb, kprojb, walb,
                                                     stats_m, stats_s);

    attn_accum<<<Bdim * 8 * 16, 256, 180224, stream>>>(qprojb, kprojb, walb,
                                                       vproj, stats_m, stats_s, vsum);

    // final projection to output: out = value_sum @ Wvo^T
    gemm_xwT<<<16, 256, 131072, stream>>>(vsum, wvob, d_out, 0);
}
